// SelfAttentionV1_29102698398388
// MI455X (gfx1250) — matrix operand order repaired and checked by reading, not compiled, not run on hardware
//
#include <hip/hip_runtime.h>
#include <hip/hip_bf16.h>

typedef __bf16 bf16;
typedef __attribute__((ext_vector_type(16))) __bf16 v16bf;
typedef __attribute__((ext_vector_type(8)))  __bf16 v8bf;
typedef __attribute__((ext_vector_type(8)))  float  v8f;

#define SEQ_N 8192
#define DIM_D 1024
#define APAD  40   // panel row pitch in bf16 elements (32 data + 8 pad, keeps 16B align)

// ---------- bf16 helpers (round-to-nearest-even) ----------
static __device__ __forceinline__ bf16 to_bf16(float f) {
  unsigned u = __builtin_bit_cast(unsigned, f);
  unsigned r = u + 0x7fffu + ((u >> 16) & 1u);
  unsigned short h = (unsigned short)(r >> 16);
  return __builtin_bit_cast(bf16, h);
}
static __device__ __forceinline__ bf16 to_bf16(bf16 b) { return b; }
static __device__ __forceinline__ float bf16_to_f32(bf16 b) {
  unsigned short h = __builtin_bit_cast(unsigned short, b);
  return __builtin_bit_cast(float, (unsigned)h << 16);
}

static __device__ __forceinline__ v16bf combine16(v8bf lo, v8bf hi) {
  v16bf r;
#pragma unroll
  for (int i = 0; i < 8; ++i) { r[i] = lo[i]; r[i + 8] = hi[i]; }
  return r;
}

static __device__ __forceinline__ v8f wmma_bf16(v16bf a, v16bf b, v8f c) {
  return __builtin_amdgcn_wmma_f32_16x16x32_bf16(
      /*neg_a=*/false, a, /*neg_b=*/false, b,
      /*c_mod=*/(short)0, c, /*reuse_a=*/false, /*reuse_b=*/false);
}

// ---------- CDNA5 async memory->LDS copy (ASYNCcnt path) ----------
// LDS aperture: flat_addr[31:0] is the LDS byte address (ISA 10.2), so the
// truncated generic pointer is a valid VDST operand. INST_OFFSET applies to
// both the LDS and global addresses, so one address pair serves a 32B copy.
static __device__ __forceinline__ unsigned lds_off(const void* p) {
  return (unsigned)(unsigned long long)p;
}
static __device__ __forceinline__ void async_copy32(unsigned lds, const void* g) {
  asm volatile("global_load_async_to_lds_b128 %0, %1, off\n\t"
               "global_load_async_to_lds_b128 %0, %1, off offset:16"
               :: "v"(lds), "v"(g) : "memory");
}
static __device__ __forceinline__ void wait_async0() {
  asm volatile("s_wait_asynccnt 0x0" ::: "memory");
}

// ---------- elementwise f32 -> bf16 ----------
__global__ __launch_bounds__(256) void cvt_f32_bf16_kernel(const float* __restrict__ in,
                                                           bf16* __restrict__ out, long n) {
  long i = (long)blockIdx.x * blockDim.x + threadIdx.x;
  long stride = (long)gridDim.x * blockDim.x;
  for (; i < n; i += stride) out[i] = to_bf16(in[i]);
}

// ---------- LDS 32x32 tile transpose: in[R,C] -> out[C,R] (bf16 out) ----------
template <typename TIN>
__global__ __launch_bounds__(256) void transpose_to_bf16_kernel(const TIN* __restrict__ in,
                                                                bf16* __restrict__ out,
                                                                int R, int C) {
  __shared__ bf16 tile[32][33];
  const int tx = threadIdx.x & 31;
  const int ty = threadIdx.x >> 5;          // 0..7
  const int c0 = blockIdx.x * 32;
  const int r0 = blockIdx.y * 32;
#pragma unroll
  for (int i = 0; i < 32; i += 8)
    tile[ty + i][tx] = to_bf16(in[(long)(r0 + ty + i) * C + (c0 + tx)]);
  __syncthreads();
#pragma unroll
  for (int i = 0; i < 32; i += 8)
    out[(long)(c0 + ty + i) * R + (r0 + tx)] = tile[tx][ty + i];
}

// ---------- GEMM: C[M,N] = scale * (A[M,K] x Bt[N,K]^T), bf16 in, f32 acc ----------
// Workgroup: 256 threads = 8 waves. Workgroup tile 128x128, double-buffered LDS
// panels (128x32 each) filled by async global->LDS copies. Wave tile 64x32
// -> 8 WMMAs per K-step.
template <bool OUT_BF16>
__global__ __launch_bounds__(256) void gemm_abt_kernel(const bf16* __restrict__ A,
                                                       const bf16* __restrict__ Bt,
                                                       void* __restrict__ Cout,
                                                       int M, int N, int K, float scale) {
  __shared__ bf16 Ap[2][128 * APAD];
  __shared__ bf16 Bp[2][128 * APAD];

  const int tid = threadIdx.x;
  const int lane = tid & 31;
  const int wave = tid >> 5;
  const int wm = wave & 1;            // 0..1 -> 64-row slab of the 128-row tile
  const int wn = wave >> 1;           // 0..3 -> 32-col slab of the 128-col tile
  const int m0 = blockIdx.x * 128;
  const int n0 = blockIdx.y * 128;

  // Staging: each thread owns one 32B half-row of each panel per K-step.
  const int srow = tid >> 1;          // 0..127
  const int soff = (tid & 1) * 16;    // 0 or 16 (bf16 elems)
  const bf16* gA = A + (long)(m0 + srow) * K + soff;
  const bf16* gB = Bt + (long)(n0 + srow) * K + soff;
  const int sidx = srow * APAD + soff;
  const unsigned ldsA[2] = { lds_off(&Ap[0][sidx]), lds_off(&Ap[1][sidx]) };
  const unsigned ldsB[2] = { lds_off(&Bp[0][sidx]), lds_off(&Bp[1][sidx]) };

  // LDS fragment bases per the WMMA lane layouts (ISA 7.12.2):
  //  A (16x32): lane&15 = M row; half-wave selects K 0-7/16-23 vs 8-15/24-31.
  //  B (32x16): lane&15 = N col; half-wave selects K 0-7/16-23 vs 8-15/24-31.
  const int a_base = (wm * 64 + (lane & 15)) * APAD + (lane >> 4) * 8;
  const int b_base = (wn * 32 + (lane & 15)) * APAD + (lane >> 4) * 8;

  v8f acc[4][2] = {};

  // preload panel 0 (async DMA into LDS, no VGPR staging)
  async_copy32(ldsA[0], gA);
  async_copy32(ldsB[0], gB);
  wait_async0();
  __syncthreads();

  int cur = 0;
  for (int kk = 0; kk < K; kk += 32) {
    // kick off next panel into the other buffer while compute consumes current
    if (kk + 32 < K) {
      async_copy32(ldsA[cur ^ 1], gA + kk + 32);
      async_copy32(ldsB[cur ^ 1], gB + kk + 32);
      __builtin_prefetch(gA + kk + 64, 0, 1);   // global_prefetch_b8 next-next
      __builtin_prefetch(gB + kk + 64, 0, 1);
    }

    // fragments from LDS (ds_load_b128 pairs)
    const bf16* ap = &Ap[cur][0] + a_base;
    const bf16* bp = &Bp[cur][0] + b_base;
    v16bf af[4], bfr[2];
#pragma unroll
    for (int j = 0; j < 2; ++j) {
      const bf16* q = bp + j * 16 * APAD;
      bfr[j] = combine16(*(const v8bf*)(q), *(const v8bf*)(q + 16));
    }
#pragma unroll
    for (int i = 0; i < 4; ++i)
      af[i] = combine16(*(const v8bf*)(ap + i * 16 * APAD),
                        *(const v8bf*)(ap + i * 16 * APAD + 16));

#pragma unroll
    for (int i = 0; i < 4; ++i)
#pragma unroll
      for (int j = 0; j < 2; ++j)
        acc[i][j] = wmma_bf16(af[i], bfr[j], acc[i][j]);

    wait_async0();     // our async copies into buf[nxt] have landed in LDS
    __syncthreads();   // all waves done reading cur / writing nxt
    cur ^= 1;
  }

  // C/D layout: lane&15 = N col; VGPR r -> M = r + (lane>>4)*8.
  const int ccol = lane & 15;
  const int rhi = (lane >> 4) * 8;
#pragma unroll
  for (int i = 0; i < 4; ++i) {
#pragma unroll
    for (int j = 0; j < 2; ++j) {
#pragma unroll
      for (int r = 0; r < 8; ++r) {
        long row = m0 + wm * 64 + i * 16 + rhi + r;
        long col = n0 + wn * 32 + j * 16 + ccol;
        float val = acc[i][j][r] * scale;
        if constexpr (OUT_BF16)
          ((bf16*)Cout)[row * (long)N + col] = to_bf16(val);
        else
          ((float*)Cout)[row * (long)N + col] = val;
      }
    }
  }
}

// ---------- row softmax in place on bf16 S[8192, 8192] ----------
__global__ __launch_bounds__(256) void softmax_rows_kernel(bf16* __restrict__ S) {
  constexpr int PER = SEQ_N / 256;           // 32 values per thread, kept in VGPRs
  __shared__ float red[256];
  const int t = threadIdx.x;
  bf16* p = S + (long)blockIdx.x * SEQ_N;

  float v[PER];
  float mx = -3.402823466e38f;
#pragma unroll
  for (int i = 0; i < PER; ++i) {
    v[i] = bf16_to_f32(p[t + i * 256]);
    mx = fmaxf(mx, v[i]);
  }
  red[t] = mx;
  __syncthreads();
  for (int s = 128; s > 0; s >>= 1) {
    if (t < s) red[t] = fmaxf(red[t], red[t + s]);
    __syncthreads();
  }
  mx = red[0];
  __syncthreads();

  float sum = 0.0f;
#pragma unroll
  for (int i = 0; i < PER; ++i) {
    v[i] = __expf(v[i] - mx);
    sum += v[i];
  }
  red[t] = sum;
  __syncthreads();
  for (int s = 128; s > 0; s >>= 1) {
    if (t < s) red[t] += red[t + s];
    __syncthreads();
  }
  const float inv = 1.0f / red[0];
#pragma unroll
  for (int i = 0; i < PER; ++i) p[t + i * 256] = to_bf16(v[i] * inv);
}

// ---------- host orchestration ----------
extern "C" void kernel_launch(void* const* d_in, const int* in_sizes, int n_in,
                              void* d_out, int out_size, void* d_ws, size_t ws_size,
                              hipStream_t stream) {
  (void)in_sizes; (void)n_in; (void)out_size; (void)ws_size;
  const int Nn = SEQ_N, D = DIM_D;

  const float* x  = (const float*)d_in[0];
  const float* Wq = (const float*)d_in[1];
  const float* Wk = (const float*)d_in[2];
  const float* Wv = (const float*)d_in[3];
  float* out = (float*)d_out;

  // workspace carve-up (256B aligned)
  char* ws = (char*)d_ws;
  size_t off = 0;
  auto take = [&](size_t bytes) -> char* {
    char* p = ws + off;
    off += (bytes + 255) & ~(size_t)255;
    return p;
  };
  bf16* xb  = (bf16*)take((size_t)Nn * D * 2);
  bf16* wqt = (bf16*)take((size_t)D * D * 2);
  bf16* wkt = (bf16*)take((size_t)D * D * 2);
  bf16* wvt = (bf16*)take((size_t)D * D * 2);
  bf16* qb  = (bf16*)take((size_t)Nn * D * 2);
  bf16* kb  = (bf16*)take((size_t)Nn * D * 2);
  bf16* vb  = (bf16*)take((size_t)Nn * D * 2);
  bf16* vtb = (bf16*)take((size_t)Nn * D * 2);   // v transposed [D, N]
  bf16* Sb  = (bf16*)take((size_t)Nn * Nn * 2);  // scores / probabilities

  // 1) x -> bf16 ; W -> bf16, transposed to [out,in] so GEMM B operand is [N,K]
  cvt_f32_bf16_kernel<<<2048, 256, 0, stream>>>(x, xb, (long)Nn * D);
  dim3 gW(D / 32, D / 32);
  transpose_to_bf16_kernel<float><<<gW, 256, 0, stream>>>(Wq, wqt, D, D);
  transpose_to_bf16_kernel<float><<<gW, 256, 0, stream>>>(Wk, wkt, D, D);
  transpose_to_bf16_kernel<float><<<gW, 256, 0, stream>>>(Wv, wvt, D, D);

  // 2) projections (reference name-swap: q = x@Wk, k = x@Wq, v = x@Wv)
  dim3 gProj(Nn / 128, D / 128);
  gemm_abt_kernel<true><<<gProj, 256, 0, stream>>>(xb, wkt, qb, Nn, D, D, 1.0f);
  gemm_abt_kernel<true><<<gProj, 256, 0, stream>>>(xb, wqt, kb, Nn, D, D, 1.0f);
  gemm_abt_kernel<true><<<gProj, 256, 0, stream>>>(xb, wvt, vb, Nn, D, D, 1.0f);

  // 3) transpose v -> [D, N] so out-GEMM uses the contiguous Bt path
  dim3 gVT(D / 32, Nn / 32);
  transpose_to_bf16_kernel<bf16><<<gVT, 256, 0, stream>>>(vb, vtb, Nn, D);

  // 4) scores S = (q @ k^T) / 32   (k is already [N,K] = k^T's transpose)
  dim3 gS(Nn / 128, Nn / 128);
  gemm_abt_kernel<true><<<gS, 256, 0, stream>>>(qb, kb, Sb, Nn, Nn, D, 0.03125f);

  // 5) row softmax in place (f32 math, bf16 storage)
  softmax_rows_kernel<<<Nn, 256, 0, stream>>>(Sb);

  // 6) out = P @ v  (A = P [N, N], Bt = v^T [D, N]), f32 output
  dim3 gO(Nn / 128, D / 128);
  gemm_abt_kernel<false><<<gO, 256, 0, stream>>>(Sb, vtb, out, Nn, D, Nn, 1.0f);
}
